// SiamAttnBlock_8143257993868
// MI455X (gfx1250) — matrix operand order repaired and checked by reading, not compiled, not run on hardware
//
#include <hip/hip_runtime.h>
#include <hip/hip_bf16.h>

// ---------------------------------------------------------------------------
// SiamAttn block for MI455X (gfx1250, wave32, WMMA).
// Matrix math on v_wmma_f32_16x16x32_bf16 (f32 accumulate), softmax in f32.
// GEMM staging uses GLOBAL_LOAD_ASYNC_TO_LDS_B128 (ASYNCcnt) with explicit,
// manually two-staged double buffering + forced unroll so the direct
// global->LDS DMA path overlaps a fully unrolled WMMA stream.
// ---------------------------------------------------------------------------

typedef __attribute__((ext_vector_type(16))) __bf16 bfx16;
typedef __attribute__((ext_vector_type(8)))  float  fx8;

#define LOG2E 1.4426950408889634f

__device__ __forceinline__ unsigned short f2bf(float f) {
  unsigned int u = __float_as_uint(f);
  u += 0x7fffu + ((u >> 16) & 1u);   // round-to-nearest-even
  return (unsigned short)(u >> 16);
}

__device__ __forceinline__ fx8 zero8() {
  fx8 z;
  #pragma unroll
  for (int i = 0; i < 8; ++i) z[i] = 0.0f;
  return z;
}

union ABFrag {             // 16 bf16 = 32 bytes = one WMMA A/B operand per lane
  bfx16 v;
  uint4 q[2];
};

// 16B global -> LDS direct async copy (per active lane); tracked by ASYNCcnt.
// volatile asms keep program order among themselves; the ordering vs LDS reads
// is enforced by s_wait_asynccnt(memory) + __syncthreads().
__device__ __forceinline__ void async_cp16(unsigned lds_off,
                                           const unsigned short* g) {
  unsigned long long ga = (unsigned long long)g;
  asm volatile("global_load_async_to_lds_b128 %0, %1, off"
               :: "v"(lds_off), "v"(ga));
}

// ---------------------------------------------------------------------------
// Elementwise conversion kernels
// ---------------------------------------------------------------------------
__global__ __launch_bounds__(256) void cvt_f32_bf16_kernel(
    const float* __restrict__ src, unsigned short* __restrict__ dst, int n) {
  int i = blockIdx.x * blockDim.x + threadIdx.x;
  int stride = gridDim.x * blockDim.x;
  for (; i < n; i += stride) dst[i] = f2bf(src[i]);
}

// xb[b, s(0..2047), h] : s<1024 from x, s>=1024 from x2 (cross-branch concat)
__global__ __launch_bounds__(256) void combine_x_kernel(
    const float* __restrict__ x, const float* __restrict__ x2,
    unsigned short* __restrict__ xb) {
  const int n = 2 * 2048 * 1024;
  int i = blockIdx.x * blockDim.x + threadIdx.x;
  int stride = gridDim.x * blockDim.x;
  for (; i < n; i += stride) {
    int hcol = i & 1023;
    int s    = (i >> 10) & 2047;
    int b    = i >> 21;
    float v = (s < 1024) ? x[((size_t)b * 1024 + s) * 1024 + hcol]
                         : x2[((size_t)b * 1024 + (s - 1024)) * 1024 + hcol];
    xb[i] = f2bf(v);
  }
}

// ---------------------------------------------------------------------------
// Shared GEMM mainloop:  C[128x128] = A[M,K] * W[N,K]^T   (both K-contiguous)
// 256 threads = 8 waves; wave grid 4(M) x 2(N); each wave -> 32x64 of C
// (2x4 WMMA tiles).  Double-buffered 128x32 bf16 LDS tiles (padded stride 40)
// filled with global_load_async_to_lds_b128; two slabs per iteration with
// compile-time buffer offsets; k-loop force-unrolled.
// ---------------------------------------------------------------------------
#define LDS_STRIDE 40
#define SLAB (128 * LDS_STRIDE)     // elements per tile buffer

__device__ __forceinline__ void gemm_mainloop(
    const unsigned short* __restrict__ A, int lda,
    const unsigned short* __restrict__ W, int ldw,
    int K, int rowA0, int rowW0,
    unsigned short* laA,            // [2][SLAB]
    unsigned short* laB,            // [2][SLAB]
    fx8 acc[2][4]) {
  const int tid  = threadIdx.x;
  const int lane = tid & 31;
  const int wave = tid >> 5;
  const int wm   = wave & 3;        // M direction, 32 rows per wave
  const int wn   = wave >> 2;       // N direction, 64 cols per wave
  const int half = lane >> 4;
  const int l16  = lane & 15;

  // per-thread staging coordinates: 2 x 16B for A, 2 x 16B for B per slab
  const int r0 = tid >> 2, r1 = r0 + 64;
  const int kc = (tid & 3) << 3;
  const unsigned ldsA = (unsigned)(size_t)laA;   // LDS byte address (addr[31:0])
  const unsigned ldsB = (unsigned)(size_t)laB;

  #pragma unroll
  for (int mt = 0; mt < 2; ++mt)
    #pragma unroll
    for (int nt = 0; nt < 4; ++nt) acc[mt][nt] = zero8();

  auto stage = [&](int k0, unsigned bofs) {
    async_cp16(ldsA + bofs + (unsigned)((r0 * LDS_STRIDE + kc) * 2),
               A + (size_t)(rowA0 + r0) * lda + k0 + kc);
    async_cp16(ldsA + bofs + (unsigned)((r1 * LDS_STRIDE + kc) * 2),
               A + (size_t)(rowA0 + r1) * lda + k0 + kc);
    async_cp16(ldsB + bofs + (unsigned)((r0 * LDS_STRIDE + kc) * 2),
               W + (size_t)(rowW0 + r0) * ldw + k0 + kc);
    async_cp16(ldsB + bofs + (unsigned)((r1 * LDS_STRIDE + kc) * 2),
               W + (size_t)(rowW0 + r1) * ldw + k0 + kc);
  };

  auto compute = [&](const unsigned short* bA, const unsigned short* bB) {
    // fragments (ISA 16-bit A 16x32 / B 32x16 lane layouts)
    ABFrag afr[2], bfr[4];
    #pragma unroll
    for (int mt = 0; mt < 2; ++mt) {
      int m  = wm * 32 + mt * 16 + l16;
      int kb = half * 8;     // lanes 0-15: K 0-7/16-23, lanes 16-31: K 8-15/24-31
      afr[mt].q[0] = *(const uint4*)(bA + m * LDS_STRIDE + kb);
      afr[mt].q[1] = *(const uint4*)(bA + m * LDS_STRIDE + kb + 16);
    }
    #pragma unroll
    for (int nt = 0; nt < 4; ++nt) {
      int n  = wn * 64 + nt * 16 + l16;
      int kb = half * 8;     // lanes 0-15: K 0-7/16-23, lanes 16-31: K 8-15/24-31
      bfr[nt].q[0] = *(const uint4*)(bB + n * LDS_STRIDE + kb);
      bfr[nt].q[1] = *(const uint4*)(bB + n * LDS_STRIDE + kb + 16);
    }
    #pragma unroll
    for (int mt = 0; mt < 2; ++mt)
      #pragma unroll
      for (int nt = 0; nt < 4; ++nt)
        acc[mt][nt] = __builtin_amdgcn_wmma_f32_16x16x32_bf16(
            false, afr[mt].v, false, bfr[nt].v, (short)0, acc[mt][nt], false, false);
  };

  stage(0, 0u);                                   // prologue: fill buffer 0

  // K is a multiple of 64 here (1024): two slabs per iteration, fixed buffers.
  #pragma unroll
  for (int k0 = 0; k0 < K; k0 += 64) {
    // ---- slab k0 in buffer 0; overlap copy of slab k0+32 into buffer 1
    stage(k0 + 32, (unsigned)(SLAB * 2));
    // async loads complete in order: <=4 outstanding => buffer-0 slab landed
    asm volatile("s_wait_asynccnt 0x4" ::: "memory");
    __builtin_prefetch(A + (size_t)(rowA0 + (tid >> 1)) * lda + (k0 + 64), 0, 1);
    __builtin_prefetch(W + (size_t)(rowW0 + (tid >> 1)) * ldw + (k0 + 64), 0, 1);
    __syncthreads();
    compute(laA, laB);
    __syncthreads();

    // ---- slab k0+32 in buffer 1; overlap copy of slab k0+64 into buffer 0
    if (k0 + 64 < K) {
      stage(k0 + 64, 0u);
      asm volatile("s_wait_asynccnt 0x4" ::: "memory");
    } else {
      asm volatile("s_wait_asynccnt 0x0" ::: "memory");
    }
    __syncthreads();
    compute(laA + SLAB, laB + SLAB);
    __syncthreads();
  }
}

// ---------------------------------------------------------------------------
// QKV GEMM: xb[b,2048,1024] @ w_qkv[3072,1024]^T, scattered to Q/K/Vt (bf16)
// Q,K: [b*16+h, 2048, 64]   Vt: [b*16+h, 64, 2048]
// ---------------------------------------------------------------------------
__global__ __launch_bounds__(256) void qkv_gemm_kernel(
    const unsigned short* __restrict__ xb, const unsigned short* __restrict__ wq,
    unsigned short* __restrict__ Qb, unsigned short* __restrict__ Kb,
    unsigned short* __restrict__ Vt) {
  __shared__ __align__(16) unsigned short laA[2 * SLAB];
  __shared__ __align__(16) unsigned short laB[2 * SLAB];
  const int b     = blockIdx.z;
  const int rowA0 = blockIdx.x * 128;
  const int rowW0 = blockIdx.y * 128;
  fx8 acc[2][4];
  gemm_mainloop(xb + (size_t)b * 2048 * 1024, 1024, wq, 1024, 1024,
                rowA0, rowW0, laA, laB, acc);

  const int lane = threadIdx.x & 31;
  const int wave = threadIdx.x >> 5;
  const int wm = wave & 3, wn = wave >> 2;
  const int half = lane >> 4, l16 = lane & 15;
  #pragma unroll
  for (int mt = 0; mt < 2; ++mt)
    #pragma unroll
    for (int nt = 0; nt < 4; ++nt)
      #pragma unroll
      for (int r = 0; r < 8; ++r) {
        int s = rowA0 + wm * 32 + mt * 16 + half * 8 + r;    // C row (ISA layout)
        int j = rowW0 + wn * 64 + nt * 16 + l16;             // C col
        int t = j >> 10, h = (j >> 6) & 15, dd = j & 63;
        unsigned short v = f2bf(acc[mt][nt][r]);
        size_t bh = (size_t)(b * 16 + h);
        if (t == 0)      Qb[(bh * 2048 + s) * 64 + dd] = v;
        else if (t == 1) Kb[(bh * 2048 + s) * 64 + dd] = v;
        else             Vt[(bh * 64 + dd) * 2048 + s] = v;
      }
}

// ---------------------------------------------------------------------------
// Flash attention: one wave = 16 query rows x d=64; loop keys in 32-wide tiles.
// ---------------------------------------------------------------------------
__global__ __launch_bounds__(256) void attn_kernel(
    const unsigned short* __restrict__ Qb, const unsigned short* __restrict__ Kb,
    const unsigned short* __restrict__ Vt, unsigned short* __restrict__ ob) {
  __shared__ __align__(16) unsigned short plds[8][16 * LDS_STRIDE];  // per-wave P tile
  const int bh   = blockIdx.y;                 // 0..31
  const int b    = bh >> 4, h = bh & 15;
  const int wave = threadIdx.x >> 5;
  const int lane = threadIdx.x & 31;
  const int half = lane >> 4, l16 = lane & 15;
  const int q0   = blockIdx.x * 128 + wave * 16;

  const unsigned short* Q  = Qb + (size_t)bh * 2048 * 64;
  const unsigned short* Kp = Kb + (size_t)bh * 2048 * 64;
  const unsigned short* Vp = Vt + (size_t)bh * 64 * 2048;
  unsigned short* pl = &plds[wave][0];

  // Q A-fragments over d (two K=32 chunks), kept resident
  ABFrag qf[2];
  #pragma unroll
  for (int kcq = 0; kcq < 2; ++kcq) {
    int m  = q0 + l16;
    int kb = kcq * 32 + half * 8;
    qf[kcq].q[0] = *(const uint4*)(Q + (size_t)m * 64 + kb);
    qf[kcq].q[1] = *(const uint4*)(Q + (size_t)m * 64 + kb + 16);
  }

  float mrow[8], lrow[8];
  #pragma unroll
  for (int r = 0; r < 8; ++r) { mrow[r] = -1e30f; lrow[r] = 0.0f; }
  fx8 acc[4];
  #pragma unroll
  for (int nt = 0; nt < 4; ++nt) acc[nt] = zero8();

  #pragma unroll 2
  for (int y0 = 0; y0 < 2048; y0 += 32) {
    // ---- scores S[16 x 32] = Q * K^T (two 16-col C tiles), scaled by d^-0.5
    fx8 s[2];
    #pragma unroll
    for (int ct = 0; ct < 2; ++ct) {
      ABFrag kf0, kf1;
      int y  = y0 + ct * 16 + l16;             // key row for this lane (B operand N)
      int kb = half * 8;
      kf0.q[0] = *(const uint4*)(Kp + (size_t)y * 64 + kb);
      kf0.q[1] = *(const uint4*)(Kp + (size_t)y * 64 + kb + 16);
      kf1.q[0] = *(const uint4*)(Kp + (size_t)y * 64 + 32 + kb);
      kf1.q[1] = *(const uint4*)(Kp + (size_t)y * 64 + 32 + kb + 16);
      fx8 c = zero8();
      c = __builtin_amdgcn_wmma_f32_16x16x32_bf16(false, qf[0].v, false, kf0.v,
                                                  (short)0, c, false, false);
      c = __builtin_amdgcn_wmma_f32_16x16x32_bf16(false, qf[1].v, false, kf1.v,
                                                  (short)0, c, false, false);
      #pragma unroll
      for (int r = 0; r < 8; ++r) s[ct][r] = c[r] * 0.125f;   // 64^-0.5
    }

    // ---- online softmax row stats (rows live at reg r + 8*half across 16 lanes)
    float nm[8], fac[8];
    #pragma unroll
    for (int r = 0; r < 8; ++r) {
      float mx = fmaxf(s[0][r], s[1][r]);
      #pragma unroll
      for (int off = 8; off > 0; off >>= 1) mx = fmaxf(mx, __shfl_xor(mx, off));
      float newm = fmaxf(mrow[r], mx);
      fac[r] = exp2f((mrow[r] - newm) * LOG2E);
      nm[r]  = newm;
    }
    #pragma unroll
    for (int ct = 0; ct < 2; ++ct)
      #pragma unroll
      for (int r = 0; r < 8; ++r)
        s[ct][r] = exp2f((s[ct][r] - nm[r]) * LOG2E);
    #pragma unroll
    for (int r = 0; r < 8; ++r) {
      float ps = s[0][r] + s[1][r];
      #pragma unroll
      for (int off = 8; off > 0; off >>= 1) ps += __shfl_xor(ps, off);
      lrow[r] = lrow[r] * fac[r] + ps;
      mrow[r] = nm[r];
    }
    #pragma unroll
    for (int nt = 0; nt < 4; ++nt)
      #pragma unroll
      for (int r = 0; r < 8; ++r) acc[nt][r] *= fac[r];

    // ---- C-layout -> A-operand layout via LDS (bf16 P tile, 16x32, stride 40)
    __syncthreads();                            // previous iter's reads complete
    #pragma unroll
    for (int ct = 0; ct < 2; ++ct)
      #pragma unroll
      for (int r = 0; r < 8; ++r)
        pl[(half * 8 + r) * LDS_STRIDE + ct * 16 + l16] = f2bf(s[ct][r]);
    __syncthreads();

    ABFrag pf;
    {
      int kb = half * 8;
      pf.q[0] = *(const uint4*)(pl + l16 * LDS_STRIDE + kb);
      pf.q[1] = *(const uint4*)(pl + l16 * LDS_STRIDE + kb + 16);
    }

    // ---- O[16x64] += P[16x32] * V[32x64]  (B operand rows from Vt, contiguous)
    #pragma unroll
    for (int nt = 0; nt < 4; ++nt) {
      ABFrag vf;
      int n  = nt * 16 + l16;
      int kb = half * 8;
      vf.q[0] = *(const uint4*)(Vp + (size_t)n * 2048 + y0 + kb);
      vf.q[1] = *(const uint4*)(Vp + (size_t)n * 2048 + y0 + kb + 16);
      acc[nt] = __builtin_amdgcn_wmma_f32_16x16x32_bf16(
          false, pf.v, false, vf.v, (short)0, acc[nt], false, false);
    }
  }

  // ---- normalize and emit bf16 [b, s, h*64+d] for the out-proj GEMM
  #pragma unroll
  for (int r = 0; r < 8; ++r) lrow[r] = 1.0f / lrow[r];
  #pragma unroll
  for (int nt = 0; nt < 4; ++nt)
    #pragma unroll
    for (int r = 0; r < 8; ++r) {
      int srow = q0 + half * 8 + r;
      int col  = h * 64 + nt * 16 + l16;
      ob[((size_t)b * 2048 + srow) * 1024 + col] = f2bf(acc[nt][r] * lrow[r]);
    }
}

// ---------------------------------------------------------------------------
// Out projection: ob[b,2048,1024] @ w_out[1024,1024]^T + b_out, split o1 / o2
// ---------------------------------------------------------------------------
__global__ __launch_bounds__(256) void proj_gemm_kernel(
    const unsigned short* __restrict__ ob, const unsigned short* __restrict__ wo,
    const float* __restrict__ bias, float* __restrict__ out) {
  __shared__ __align__(16) unsigned short laA[2 * SLAB];
  __shared__ __align__(16) unsigned short laB[2 * SLAB];
  const int b     = blockIdx.z;
  const int rowA0 = blockIdx.x * 128;
  const int rowW0 = blockIdx.y * 128;
  fx8 acc[2][4];
  gemm_mainloop(ob + (size_t)b * 2048 * 1024, 1024, wo, 1024, 1024,
                rowA0, rowW0, laA, laB, acc);

  const int lane = threadIdx.x & 31;
  const int wave = threadIdx.x >> 5;
  const int wm = wave & 3, wn = wave >> 2;
  const int half = lane >> 4, l16 = lane & 15;
  const size_t half_out = (size_t)2 * 1024 * 1024;   // o1 flat size
  #pragma unroll
  for (int mt = 0; mt < 2; ++mt)
    #pragma unroll
    for (int nt = 0; nt < 4; ++nt)
      #pragma unroll
      for (int r = 0; r < 8; ++r) {
        int s = rowA0 + wm * 32 + mt * 16 + half * 8 + r;
        int j = rowW0 + wn * 64 + nt * 16 + l16;
        float y = acc[mt][nt][r] + bias[j];
        size_t idx = (s < 1024)
            ? ((size_t)b * 1024 + s) * 1024 + j
            : half_out + ((size_t)b * 1024 + (s - 1024)) * 1024 + j;
        out[idx] = y;
      }
}

// ---------------------------------------------------------------------------
extern "C" void kernel_launch(void* const* d_in, const int* in_sizes, int n_in,
                              void* d_out, int out_size, void* d_ws, size_t ws_size,
                              hipStream_t stream) {
  (void)in_sizes; (void)n_in; (void)out_size; (void)ws_size;
  const float* x    = (const float*)d_in[0];
  const float* x2   = (const float*)d_in[1];
  const float* wqkv = (const float*)d_in[2];   // [3072,1024]
  const float* wout = (const float*)d_in[3];   // [1024,1024]
  const float* bout = (const float*)d_in[4];   // [1024]
  // d_in[5] = num_heads (fixed 16 for these shapes)

  char* ws = (char*)d_ws;
  const size_t MB = 1024 * 1024;
  unsigned short* xb  = (unsigned short*)(ws + 0 * MB);   // 8 MiB [2,2048,1024]
  unsigned short* wqb = (unsigned short*)(ws + 8 * MB);   // 6 MiB [3072,1024]
  unsigned short* wob = (unsigned short*)(ws + 14 * MB);  // 2 MiB [1024,1024]
  unsigned short* Qb  = (unsigned short*)(ws + 16 * MB);  // 8 MiB [32,2048,64]
  unsigned short* Kb  = (unsigned short*)(ws + 24 * MB);  // 8 MiB [32,2048,64]
  unsigned short* Vt  = (unsigned short*)(ws + 32 * MB);  // 8 MiB [32,64,2048]
  unsigned short* ob  = (unsigned short*)(ws + 40 * MB);  // 8 MiB [2,2048,1024]

  combine_x_kernel<<<dim3(1024), dim3(256), 0, stream>>>(x, x2, xb);
  cvt_f32_bf16_kernel<<<dim3(768), dim3(256), 0, stream>>>(wqkv, wqb, 3072 * 1024);
  cvt_f32_bf16_kernel<<<dim3(256), dim3(256), 0, stream>>>(wout, wob, 1024 * 1024);
  qkv_gemm_kernel<<<dim3(16, 24, 2), dim3(256), 0, stream>>>(xb, wqb, Qb, Kb, Vt);
  attn_kernel<<<dim3(16, 32), dim3(256), 0, stream>>>(Qb, Kb, Vt, ob);
  proj_gemm_kernel<<<dim3(16, 8, 2), dim3(256), 0, stream>>>(ob, wob, bout, (float*)d_out);
}
